// SingleHeadAttention_10582799417886
// MI455X (gfx1250) — matrix operand order repaired and checked by reading, not compiled, not run on hardware
//
#include <hip/hip_runtime.h>
#include <hip/hip_bf16.h>
#include <math.h>

// ---------------------------------------------------------------------------
// Types for CDNA5 WMMA bf16: D(f32 16x16) = A(bf16 16x32) * B(bf16 32x16) + C
// ---------------------------------------------------------------------------
typedef __bf16 v16bf __attribute__((ext_vector_type(16)));
typedef float  v8f   __attribute__((ext_vector_type(8)));

union Frag16 {      // one 16-bit A/B operand (8 VGPRs / 16 halfs per lane)
    v16bf v;
    uint4 u[2];
};

union H8 {
    unsigned short s[8];
    uint4 u;
};

// f32 -> bf16 (round-half-up): high half of (bits + 0x8000).
__device__ __forceinline__ unsigned short bf16bits(float f) {
    unsigned int u = __builtin_bit_cast(unsigned int, f) + 0x8000u;
    return (unsigned short)(u >> 16);
}

// Pack two f32 into one bf16x2 dword: 2 adds + 1 v_perm_b32.
__device__ __forceinline__ unsigned int pk2(float a, float b) {
    unsigned int ua = __builtin_bit_cast(unsigned int, a) + 0x8000u;
    unsigned int ub = __builtin_bit_cast(unsigned int, b) + 0x8000u;
    return __builtin_amdgcn_perm(ub, ua, 0x07060302u);
}

// 8 consecutive f32 -> 8 bf16 (one uint4)
__device__ __forceinline__ uint4 cvt8(const float* p) {
    float4 a = *(const float4*)p;
    float4 b = *(const float4*)(p + 4);
    uint4 r;
    r.x = pk2(a.x, a.y);
    r.y = pk2(a.z, a.w);
    r.z = pk2(b.x, b.y);
    r.w = pk2(b.z, b.w);
    return r;
}

#define WMMA_BF16(A, B, C) \
    __builtin_amdgcn_wmma_f32_16x16x32_bf16(false, (A), false, (B), (short)0, (C), false, false)

// Problem constants
#define BATCH 8
#define SEQ   2048
#define EMB   1024
#define HD    128
#define MROWS (BATCH * SEQ)       // 16384
#define LPAD  40                  // LDS row stride in halfs (32 + 8 pad)

#if __has_builtin(__builtin_amdgcn_tensor_load_to_lds)
#define HAVE_TDM 1
typedef unsigned int u32x4 __attribute__((ext_vector_type(4)));
typedef int          i32x8 __attribute__((ext_vector_type(8)));
typedef int          i32x4 __attribute__((ext_vector_type(4)));
#endif

// ---------------------------------------------------------------------------
// Kernel 0: convert + transpose W: (E,H) f32 -> Wbf[p][h][e] bf16
// ---------------------------------------------------------------------------
__global__ void __launch_bounds__(256)
convert_w_kernel(const float* __restrict__ Wq, const float* __restrict__ Wk,
                 const float* __restrict__ Wv, unsigned short* __restrict__ Wbf)
{
    const int p = blockIdx.y;
    const float* W = (p == 0) ? Wq : (p == 1) ? Wk : Wv;
    unsigned short* dst = Wbf + (size_t)p * HD * EMB;
    const int h = blockIdx.x;          // 0..127
#pragma unroll
    for (int it = 0; it < EMB / 256; ++it) {
        int e = it * 256 + threadIdx.x;
        dst[(size_t)h * EMB + e] = bf16bits(W[(size_t)e * HD + h]);
    }
}

// ---------------------------------------------------------------------------
// Kernel 1: fused projection  P = x(16384x1024) * W(1024x128) -> bf16
// W chunk staged into LDS by the Tensor Data Mover (TDM) with row padding.
// grid.x = M/64 tiles, grid.y = 3 (q,k,v); 128 threads = 4 waves, 16 rows/wave
// ---------------------------------------------------------------------------
__global__ void __launch_bounds__(128)
proj_kernel(const float* __restrict__ x,
            const unsigned short* __restrict__ Wbf,
            unsigned short* __restrict__ qb,
            unsigned short* __restrict__ kb,
            unsigned short* __restrict__ vb)
{
    __shared__ __align__(16) unsigned short Wt[HD * LPAD];  // W chunk, [h][kk], stride LPAD

    const int p = blockIdx.y;
    const unsigned short* Wsrc = Wbf + (size_t)p * HD * EMB;  // [h][e]
    unsigned short* out = (p == 0) ? qb : (p == 1) ? kb : vb;

    const int tid  = threadIdx.x;
    const int wave = tid >> 5;
    const int lane = tid & 31;
    const int ln   = lane & 15;
    const int hi   = lane >> 4;
    const int mbase = blockIdx.x * 64 + wave * 16;

    v8f acc[8] = {};

    for (int e0 = 0; e0 < EMB; e0 += 32) {
#ifdef HAVE_TDM
        // TDM: DMA a 32x128 bf16 tile (rows h, cols e0..e0+31) into LDS,
        // padding 4 DWORDs after every 16 DWORDs -> row stride = 40 halfs = LPAD.
        if (wave == 0) {
            unsigned long long ga = (unsigned long long)(uintptr_t)(Wsrc + e0);
            unsigned int lds = (unsigned int)(uintptr_t)&Wt[0];
            u32x4 g0;
            g0.x = 1u;                                      // count=1, user mode
            g0.y = lds;                                     // lds_addr
            g0.z = (unsigned int)ga;                        // global_addr[31:0]
            g0.w = (unsigned int)((ga >> 32) & 0x01FFFFFFull) | (2u << 30); // addr[56:32], type=2
            i32x8 g1 = {
                (1 << 16) | (1 << 20) | (3 << 22) | (3 << 25), // data_size=2B, pad_en, pad_interval=16DW, pad_amount=4DW
                (int)(1024u << 16),                            // tensor_dim0 = 1024 (bits 63:48)
                (int)(128u  << 16),                            // tensor_dim1 = 128  (bits 95:80)
                (int)(32u   << 16),                            // tile_dim0   = 32   (bits 127:112)
                128,                                           // tile_dim1   = 128  (bits 143:128)
                1024,                                          // tensor_dim0_stride = 1024 (bits 191:160)
                0, 0
            };
            i32x4 z4 = { 0, 0, 0, 0 };
            i32x8 z8 = { 0, 0, 0, 0, 0, 0, 0, 0 };
            __builtin_amdgcn_tensor_load_to_lds(g0, g1, z4, z4, z8, 0);
            __builtin_amdgcn_s_wait_tensorcnt((short)0);
        }
#else
        // Fallback: cooperative copy, thread t owns row h=t (W already transposed)
        {
            const uint4* src = (const uint4*)(Wsrc + (size_t)tid * EMB + e0);
            uint4* dstrow = (uint4*)&Wt[tid * LPAD];
            dstrow[0] = src[0];
            dstrow[1] = src[1];
        }
#endif
        __syncthreads();

        // A fragment from x (f32 -> bf16 on the fly), 16x32 A layout:
        // lane(hi) half0 = K[8hi..8hi+7], half1 = K[16+8hi..23+8hi]
        Frag16 a;
        const float* xrow = x + (size_t)(mbase + ln) * EMB + e0;
        a.u[0] = cvt8(xrow + 8 * hi);
        a.u[1] = cvt8(xrow + 16 + 8 * hi);

        // Preload all 8 B fragments, then issue WMMAs back-to-back
        Frag16 bfr[8];
#pragma unroll
        for (int j = 0; j < 8; ++j) {
            const unsigned short* wrow = &Wt[(16 * j + ln) * LPAD + 8 * hi];
            bfr[j].u[0] = *(const uint4*)(wrow);
            bfr[j].u[1] = *(const uint4*)(wrow + 16);
        }
#pragma unroll
        for (int j = 0; j < 8; ++j)
            acc[j] = WMMA_BF16(a.v, bfr[j].v, acc[j]);
        __syncthreads();
    }

    // D layout: vgpr i -> row (i + 8*hi), lane ln -> col; store bf16
#pragma unroll
    for (int j = 0; j < 8; ++j)
#pragma unroll
        for (int i = 0; i < 8; ++i) {
            int m = mbase + i + 8 * hi;
            out[(size_t)m * HD + 16 * j + ln] = bf16bits(acc[j][i]);
        }
}

// ---------------------------------------------------------------------------
// Kernel 2: flash attention, one wave per 16-query tile, 32-key blocks
// grid = (C/16, B), block = 32 (wave32)
// ---------------------------------------------------------------------------
__global__ void __launch_bounds__(32)
attn_kernel(const unsigned short* __restrict__ qb,
            const unsigned short* __restrict__ kbuf,
            const unsigned short* __restrict__ vbuf,
            float* __restrict__ out)
{
    __shared__ __align__(16) unsigned short Plds[16 * LPAD];  // P tile 16x32 bf16
    __shared__ __align__(16) unsigned short Vt[HD * LPAD];    // V block transposed [h][kk]

    const int lane = threadIdx.x & 31;
    const int ln   = lane & 15;
    const int hi   = lane >> 4;
    const int qt   = blockIdx.x;       // 0..127
    const int b    = blockIdx.y;       // 0..7
    const int q0   = qt * 16;
    const int rowbase = b * SEQ + q0;

    // Q fragments: 4 K-chunks of 32 over head dim
    Frag16 qf[4];
    {
        const unsigned short* qrow = qb + (size_t)(rowbase + ln) * HD;
#pragma unroll
        for (int c = 0; c < 4; ++c) {
            qf[c].u[0] = *(const uint4*)(qrow + 32 * c + 8 * hi);
            qf[c].u[1] = *(const uint4*)(qrow + 32 * c + 16 + 8 * hi);
        }
    }

    v8f acc[8] = {};
    float mrow[8], lrow[8];
#pragma unroll
    for (int i = 0; i < 8; ++i) { mrow[i] = -INFINITY; lrow[i] = 0.0f; }

    // scores scaled into log2 domain: 1/sqrt(1024) * log2(e)
    const float sscale = (1.0f / 32.0f) * 1.44269504088896f;

    const int nkb = (q0 >> 5) + 1;
    for (int kblk = 0; kblk < nkb; ++kblk) {
        const int kbase = b * SEQ + kblk * 32;

        // Prefetch next block's K and V rows (global_prefetch_b8)
        if (kblk + 1 < nkb) {
            const unsigned short* nk = kbuf + (size_t)(kbase + 32 + lane) * HD;
            const unsigned short* nv = vbuf + (size_t)(kbase + 32 + lane) * HD;
            __builtin_prefetch(nk, 0, 3);
            __builtin_prefetch(nk + 64, 0, 3);
            __builtin_prefetch(nv, 0, 3);
            __builtin_prefetch(nv + 64, 0, 3);
        }

        // Stage V block (32 keys x 128) transposed into LDS: lane owns key row
        {
            const unsigned short* vrow = vbuf + (size_t)(kbase + lane) * HD;
#pragma unroll
            for (int h8 = 0; h8 < 16; ++h8) {
                H8 d; d.u = *(const uint4*)(vrow + h8 * 8);
#pragma unroll
                for (int t = 0; t < 8; ++t)
                    Vt[(h8 * 8 + t) * LPAD + lane] = d.s[t];
            }
        }
        __syncthreads();

        // Scores S = Q * K^T for 2 subblocks of 16 keys (K-reduction over HD)
        v8f S[2];
#pragma unroll
        for (int kn = 0; kn < 2; ++kn) {
            const unsigned short* krow = kbuf + (size_t)(kbase + 16 * kn + ln) * HD;
            Frag16 kf[4];
#pragma unroll
            for (int c = 0; c < 4; ++c) {
                kf[c].u[0] = *(const uint4*)(krow + 32 * c + 8 * hi);
                kf[c].u[1] = *(const uint4*)(krow + 32 * c + 8 * hi + 16);
            }
            v8f s = {};
#pragma unroll
            for (int c = 0; c < 4; ++c)
                s = WMMA_BF16(qf[c].v, kf[c].v, s);
            S[kn] = s;
        }

        // Scale + causal mask (D layout: row = i+8*hi, col = ln)
        const bool needmask = (kblk * 32 + 31 > q0);
#pragma unroll
        for (int kn = 0; kn < 2; ++kn)
#pragma unroll
            for (int i = 0; i < 8; ++i) {
                float sv = S[kn][i] * sscale;
                if (needmask) {
                    int key = kblk * 32 + kn * 16 + ln;
                    int qi  = q0 + i + 8 * hi;
                    if (key > qi) sv = -INFINITY;
                }
                S[kn][i] = sv;
            }

        // Online softmax: per-row max/sum via 16-lane xor reductions
        float alpha[8];
#pragma unroll
        for (int i = 0; i < 8; ++i) {
            float t = fmaxf(S[0][i], S[1][i]);
#pragma unroll
            for (int off = 1; off < 16; off <<= 1)
                t = fmaxf(t, __shfl_xor(t, off, 32));
            float mnew = fmaxf(mrow[i], t);
            alpha[i] = __builtin_amdgcn_exp2f(mrow[i] - mnew);
            mrow[i] = mnew;
            float p0 = __builtin_amdgcn_exp2f(S[0][i] - mnew);
            float p1 = __builtin_amdgcn_exp2f(S[1][i] - mnew);
            int m = i + 8 * hi;
            Plds[m * LPAD + ln]      = bf16bits(p0);
            Plds[m * LPAD + 16 + ln] = bf16bits(p1);
            float ps = p0 + p1;
#pragma unroll
            for (int off = 1; off < 16; off <<= 1)
                ps += __shfl_xor(ps, off, 32);
            lrow[i] = lrow[i] * alpha[i] + ps;
        }
        __syncthreads();

        // Rescale accumulator by alpha (row mapping identical to D layout)
#pragma unroll
        for (int j = 0; j < 8; ++j)
#pragma unroll
            for (int i = 0; i < 8; ++i)
                acc[j][i] *= alpha[i];

        // P in A layout from LDS (D->A transpose via LDS round trip)
        Frag16 pf;
        {
            const unsigned short* prow = &Plds[ln * LPAD];
            pf.u[0] = *(const uint4*)(prow + 8 * hi);
            pf.u[1] = *(const uint4*)(prow + 16 + 8 * hi);
        }

        // Preload all 8 V fragments, then acc += P(16x32) * V(32x16)
        Frag16 vf[8];
#pragma unroll
        for (int j = 0; j < 8; ++j) {
            const unsigned short* vtrow = &Vt[(16 * j + ln) * LPAD + 8 * hi];
            vf[j].u[0] = *(const uint4*)(vtrow);
            vf[j].u[1] = *(const uint4*)(vtrow + 16);
        }
#pragma unroll
        for (int j = 0; j < 8; ++j)
            acc[j] = WMMA_BF16(pf.v, vf[j].v, acc[j]);
        __syncthreads();
    }

    // Epilogue: divide by row sum, store f32
#pragma unroll
    for (int i = 0; i < 8; ++i) {
        float inv = 1.0f / lrow[i];
        int m = rowbase + i + 8 * hi;
#pragma unroll
        for (int j = 0; j < 8; ++j)
            out[(size_t)m * HD + 16 * j + ln] = acc[j][i] * inv;
    }
}

// ---------------------------------------------------------------------------
extern "C" void kernel_launch(void* const* d_in, const int* in_sizes, int n_in,
                              void* d_out, int out_size, void* d_ws, size_t ws_size,
                              hipStream_t stream) {
    const float* x  = (const float*)d_in[0];
    const float* Wq = (const float*)d_in[1];
    const float* Wk = (const float*)d_in[2];
    const float* Wv = (const float*)d_in[3];
    float* out = (float*)d_out;

    // Workspace: Q, K, V bf16 (4 MB each) + Wbf transposed (768 KB)
    unsigned short* qb  = (unsigned short*)d_ws;
    unsigned short* kb  = qb + (size_t)MROWS * HD;
    unsigned short* vb  = kb + (size_t)MROWS * HD;
    unsigned short* wbf = vb + (size_t)MROWS * HD;

    dim3 wgrid(HD, 3);
    convert_w_kernel<<<wgrid, 256, 0, stream>>>(Wq, Wk, Wv, wbf);

    dim3 pgrid(MROWS / 64, 3);
    proj_kernel<<<pgrid, 128, 0, stream>>>(x, wbf, qb, kb, vb);

    dim3 agrid(SEQ / 16, BATCH);
    attn_kernel<<<agrid, 32, 0, stream>>>(qb, kb, vb, out);
}
